// IPGNO_61177514164377
// MI455X (gfx1250) — matrix operand order repaired and checked by reading, not compiled, not run on hardware
//
#include <hip/hip_runtime.h>
#include <hip/hip_bf16.h>

typedef __attribute__((ext_vector_type(16))) _Float16 v16h;
typedef __attribute__((ext_vector_type(8)))  _Float16 v8h;
typedef __attribute__((ext_vector_type(8)))  float    v8f;

// Problem constants (from reference)
constexpr int Nn = 50000;    // nodes (divisible by 16)
constexpr int Ee = 1600000;  // edges (divisible by 16)
constexpr int Gg = 64;       // graphs
constexpr int Tt = 5;        // timesteps
constexpr int Hh = 64;       // hidden
constexpr int EDd = 9;       // edge dim
constexpr int EAP = 16;      // ea padded width (f16)
constexpr int IN_D = 10;
constexpr int OUT_D = 3;
constexpr float EPS = 1e-6f;

static __device__ __forceinline__ v8f wmma_f16(v16h a, v16h b, v8f c) {
  return __builtin_amdgcn_wmma_f32_16x16x32_f16(
      /*neg_a=*/false, a, /*neg_b=*/false, b,
      /*c_mod=*/(short)0, c, /*reuse_a=*/false, /*reuse_b=*/false);
}

static __device__ __forceinline__ float relu1(float x) {
  return fmaxf(x, 0.0f);
}

static __device__ __forceinline__ v16h cat8(v8h lo, v8h hi) {
  return __builtin_shufflevector(lo, hi, 0, 1, 2, 3, 4, 5, 6, 7,
                                 8, 9, 10, 11, 12, 13, 14, 15);
}

// Load a pre-transformed A-operand fragment (weights) from fragment-major LDS:
// frag[f][lane][16 contiguous f16]  ->  exactly two ds_load_b128 per fragment.
static __device__ __forceinline__ v16h load_afrag(const _Float16* frag, int f, int lane) {
  const v8h* p = (const v8h*)(frag + ((f * 32 + lane) << 4));
  return cat8(p[0], p[1]);
}

// B operand from a contiguous row of 16 f16 (16B-aligned) -> two b128 loads.
static __device__ __forceinline__ v16h load_brow(const _Float16* p) {
  const v8h* q = (const v8h*)p;
  return cat8(q[0], q[1]);
}

// Convert-then-ReLU: cvt_pk_f16_f32 output is canonical, so the packed f16
// max needs no canonicalize -> 4x v_cvt_pk + 4x v_pk_max_num_f16 total.
// (RNE conversion is monotone and 0->0, so relu(cvt(x)) == cvt(relu(x)).)
static __device__ __forceinline__ v8h relu_h8(v8f c) {
  v8h h;
#pragma unroll
  for (int j = 0; j < 8; ++j) h[j] = (_Float16)c[j];
  v8h z = {};
  return __builtin_elementwise_max(h, z);
}

static __device__ __forceinline__ v8h shflx16(v8h v) {
  union { v8h h; int i[4]; } u;
  u.h = v;
#pragma unroll
  for (int k = 0; k < 4; ++k) u.i[k] = __shfl_xor(u.i[k], 16, 32);
  return u.h;
}

// Build next-layer B fragment for one K-step from the two accumulator tiles
// cA (channels tile 2s) and cB (tile 2s+1), passed with COMPILE-TIME indices.
// Lane l and l+16 hold the same row with opposite channel halves; route with
// exactly one shuffle and three 4-dword selects (no dynamic register indexing):
//   lane<16 needs: lo = own cA half, up = partner's cA half
//   lane>=16 needs: lo = partner's cB half, up = own cB half
static __device__ __forceinline__ v16h make_bfrag(v8f cA, v8f cB, bool hi) {
  const v8h hA = relu_h8(cA);
  const v8h hB = relu_h8(cB);
  const v8h send = hi ? hA : hB;   // what the partner lane needs from us
  const v8h recv = shflx16(send);
  const v8h lo = hi ? recv : hA;
  const v8h up = hi ? hB : recv;
  return cat8(lo, up);
}

// Prologue helper: fill fragment-major LDS from row-major f32 W[K][64].
// Fragment f = nt*nk + s; A-layout element j <-> K = s*32 + koffA + j,
// row m = nt*16 + (lane&15), koffA = (lane&16)?16:0.
static __device__ __forceinline__ void fill_frags(_Float16* frag, const float* W,
                                                  int nk, int nfrag, int Kreal) {
  for (int i = threadIdx.x; i < nfrag * 512; i += 256) {
    const int f = i >> 9, L = (i >> 4) & 31, j = i & 15;
    const int nt = f / nk, s = f - nt * nk;
    const int m = nt * 16 + (L & 15);
    const int k = s * 32 + ((L & 16) ? 16 : 0) + j;
    frag[i] = (k < Kreal) ? (_Float16)W[k * 64 + m] : (_Float16)0.0f;
  }
}

// ---------------------------------------------------------------------------
// Per-graph edge_attr statistics (sum, sumsq, count) with LDS combining.
// ---------------------------------------------------------------------------
__global__ __launch_bounds__(256) void edge_stats_kernel(
    const float* __restrict__ edge_attr, const int* __restrict__ src,
    const int* __restrict__ batch,
    float* __restrict__ sums, float* __restrict__ sqs, float* __restrict__ cnt) {
  __shared__ float ls[Gg * EDd];
  __shared__ float lq[Gg * EDd];
  __shared__ float lc[Gg];
  for (int i = threadIdx.x; i < Gg * EDd; i += 256) { ls[i] = 0.f; lq[i] = 0.f; }
  if (threadIdx.x < Gg) lc[threadIdx.x] = 0.f;
  __syncthreads();

  const int stride = gridDim.x * blockDim.x;
  for (int e = blockIdx.x * blockDim.x + threadIdx.x; e < Ee; e += stride) {
    const int g = batch[src[e]];
    atomicAdd(&lc[g], 1.0f);
#pragma unroll
    for (int d = 0; d < EDd; ++d) {
      const float a = edge_attr[(size_t)e * EDd + d];
      atomicAdd(&ls[g * EDd + d], a);
      atomicAdd(&lq[g * EDd + d], a * a);
    }
  }
  __syncthreads();
  for (int i = threadIdx.x; i < Gg * EDd; i += 256) {
    atomicAdd(&sums[i], ls[i]);
    atomicAdd(&sqs[i], lq[i]);
  }
  if (threadIdx.x < Gg) atomicAdd(&cnt[threadIdx.x], lc[threadIdx.x]);
}

__global__ void stats_finalize_kernel(const float* __restrict__ sums,
                                      const float* __restrict__ sqs,
                                      const float* __restrict__ cnt,
                                      float* __restrict__ mu,
                                      float* __restrict__ inv) {
  const int i = threadIdx.x;
  if (i >= Gg * EDd) return;
  const float c = fmaxf(cnt[i / EDd], 1.0f);
  const float m = sums[i] / c;
  const float var = fmaxf(sqs[i] / c - m * m, 0.0f);
  mu[i] = m;
  inv[i] = 1.0f / (sqrtf(var) + EPS);
}

// Normalized edge attrs, padded to 16 f16 per edge so layer-0's tail K-step
// is two aligned b128 loads.
__global__ __launch_bounds__(256) void edge_normalize_kernel(
    const float* __restrict__ edge_attr, const int* __restrict__ src,
    const int* __restrict__ batch, const float* __restrict__ mu,
    const float* __restrict__ inv, _Float16* __restrict__ eaPad) {
  const int e = blockIdx.x * blockDim.x + threadIdx.x;
  if (e >= Ee) return;
  const int g = batch[src[e]];
  v8h out0, out1;
#pragma unroll
  for (int d = 0; d < 8; ++d) {
    const float a = edge_attr[(size_t)e * EDd + d];
    out0[d] = (_Float16)((a - mu[g * EDd + d]) * inv[g * EDd + d]);
  }
  {
    const float a = edge_attr[(size_t)e * EDd + 8];
    out1[0] = (_Float16)((a - mu[g * EDd + 8]) * inv[g * EDd + 8]);
  }
#pragma unroll
  for (int d = 1; d < 8; ++d) out1[d] = (_Float16)0.0f;
  v8h* dst = (v8h*)(eaPad + (size_t)e * EAP);
  dst[0] = out0;
  dst[1] = out1;
}

__global__ __launch_bounds__(256) void degree_kernel(const int* __restrict__ dst,
                                                     float* __restrict__ deg) {
  const int e = blockIdx.x * blockDim.x + threadIdx.x;
  if (e < Ee) atomicAdd(&deg[dst[e]], 1.0f);
}

// ---------------------------------------------------------------------------
// Lift: v0 = relu(x @ W1 + b1) @ W2 + b2  (computed transposed: D = W^T X^T)
// ---------------------------------------------------------------------------
__global__ __launch_bounds__(256) void lift_kernel(
    const float* __restrict__ x, const float* __restrict__ W1,
    const float* __restrict__ b1, const float* __restrict__ W2,
    const float* __restrict__ b2, _Float16* __restrict__ v16,
    _Float16* __restrict__ v0_16) {
  __shared__ _Float16 fW1[4 * 512];   // 4 nt x 1 kstep (K=10 pad 32)
  __shared__ _Float16 fW2[8 * 512];   // 4 nt x 2 ksteps
  __shared__ float sb1[64], sb2[64];

  fill_frags(fW1, W1, 1, 4, IN_D);
  fill_frags(fW2, W2, 2, 8, 64);
  if (threadIdx.x < 64) { sb1[threadIdx.x] = b1[threadIdx.x]; sb2[threadIdx.x] = b2[threadIdx.x]; }
  __syncthreads();

  const int lane = threadIdx.x & 31;
  const int wave = threadIdx.x >> 5;
  const int r = lane & 15;                  // node within tile = column of D
  const bool hi = (lane & 16) != 0;
  const int rowoff = hi ? 8 : 0;            // channel half held by this lane
  const int koffB = hi ? 16 : 0;

  v8f bias1[4], bias2[4];
#pragma unroll
  for (int nt = 0; nt < 4; ++nt)
#pragma unroll
    for (int j = 0; j < 8; ++j) {
      bias1[nt][j] = sb1[nt * 16 + rowoff + j];
      bias2[nt][j] = sb2[nt * 16 + rowoff + j];
    }

  const int nTiles = Nn / 16;
  for (int tile = blockIdx.x * 8 + wave; tile < nTiles; tile += gridDim.x * 8) {
    const size_t node = (size_t)(tile * 16 + r);
    const float* xr = x + node * IN_D;

    // layer 1: B = x^T (K=10 padded to 32); hi lanes carry only padding.
    v16h bx;
#pragma unroll
    for (int j = 0; j < 16; ++j) {
      const int K = koffB + j;
      bx[j] = (K < IN_D) ? (_Float16)xr[K] : (_Float16)0.0f;
    }
    v8f c1[4];
#pragma unroll
    for (int nt = 0; nt < 4; ++nt) {
      c1[nt] = bias1[nt];
      c1[nt] = wmma_f16(load_afrag(fW1, nt, lane), bx, c1[nt]);
    }

    // layer 2
    v8f c2[4];
#pragma unroll
    for (int nt = 0; nt < 4; ++nt) c2[nt] = bias2[nt];
#pragma unroll
    for (int s = 0; s < 2; ++s) {
      const v16h b = make_bfrag(c1[2 * s], c1[2 * s + 1], hi);
#pragma unroll
      for (int nt = 0; nt < 4; ++nt)
        c2[nt] = wmma_f16(load_afrag(fW2, nt * 2 + s, lane), b, c2[nt]);
    }

    // store: lane owns node `node`, channels nt*16+rowoff+0..7 (contiguous).
#pragma unroll
    for (int nt = 0; nt < 4; ++nt) {
      v8h out;
#pragma unroll
      for (int j = 0; j < 8; ++j) out[j] = (_Float16)c2[nt][j];
      *(v8h*)(v16 + node * 64 + nt * 16 + rowoff) = out;
      *(v8h*)(v0_16 + node * 64 + nt * 16 + rowoff) = out;
    }
  }
}

// ---------------------------------------------------------------------------
// Edge MLP (dominant): m = MLP3([v[src] | ea]); scatter-add into agg.
// Transposed form: each layer D = W^T X^T; lane = edge row, VGPRs = channels.
// ---------------------------------------------------------------------------
__global__ __launch_bounds__(256) void edge_mlp_kernel(
    const _Float16* __restrict__ vcur, const _Float16* __restrict__ eaPad,
    const int* __restrict__ src, const int* __restrict__ dst,
    const float* __restrict__ W0, const float* __restrict__ b0,
    const float* __restrict__ W1, const float* __restrict__ b1,
    const float* __restrict__ W2, const float* __restrict__ b2,
    float* __restrict__ agg) {
  __shared__ _Float16 fW0[12 * 512];  // 4 nt x 3 ksteps (K=73 pad 96)
  __shared__ _Float16 fW1[8 * 512];
  __shared__ _Float16 fW2[8 * 512];
  __shared__ float sb0[64], sb1[64], sb2[64];

  fill_frags(fW0, W0, 3, 12, 73);
  fill_frags(fW1, W1, 2, 8, 64);
  fill_frags(fW2, W2, 2, 8, 64);
  if (threadIdx.x < 64) { sb0[threadIdx.x] = b0[threadIdx.x]; sb1[threadIdx.x] = b1[threadIdx.x]; sb2[threadIdx.x] = b2[threadIdx.x]; }
  __syncthreads();

  const int lane = threadIdx.x & 31;
  const int wave = threadIdx.x >> 5;
  const int r = lane & 15;
  const bool hi = (lane & 16) != 0;
  const int rowoff = hi ? 8 : 0;
  const int koffB = hi ? 16 : 0;

  v8f bias0[4], bias1[4], bias2[4];
#pragma unroll
  for (int nt = 0; nt < 4; ++nt)
#pragma unroll
    for (int j = 0; j < 8; ++j) {
      bias0[nt][j] = sb0[nt * 16 + rowoff + j];
      bias1[nt][j] = sb1[nt * 16 + rowoff + j];
      bias2[nt][j] = sb2[nt * 16 + rowoff + j];
    }

  const int nTiles = Ee / 16;
  for (int tile = blockIdx.x * 8 + wave; tile < nTiles; tile += gridDim.x * 8) {
    const int e = tile * 16 + r;
    const _Float16* vrow = vcur + (size_t)src[e] * 64;   // 16B-aligned gather
    const _Float16* earow = eaPad + (size_t)e * EAP;

    // ---- layer 0: K = 96 (64 feat + 16 ea-pad + 16 zero)
    v8f c0[4];
#pragma unroll
    for (int nt = 0; nt < 4; ++nt) c0[nt] = bias0[nt];
#pragma unroll
    for (int s = 0; s < 3; ++s) {
      v16h b;
      if (s < 2) {
        b = load_brow(vrow + s * 32 + koffB);            // 2x global_load_b128
      } else if (!hi) {
        b = load_brow(earow);                            // features 64..79
      } else {
#pragma unroll
        for (int j = 0; j < 16; ++j) b[j] = (_Float16)0.0f;  // features 80..95
      }
#pragma unroll
      for (int nt = 0; nt < 4; ++nt)
        c0[nt] = wmma_f16(load_afrag(fW0, nt * 3 + s, lane), b, c0[nt]);
    }

    // ---- layer 1 (relu + half-swap exchange feeds B operand directly)
    v8f c1[4];
#pragma unroll
    for (int nt = 0; nt < 4; ++nt) c1[nt] = bias1[nt];
#pragma unroll
    for (int s = 0; s < 2; ++s) {
      const v16h b = make_bfrag(c0[2 * s], c0[2 * s + 1], hi);
#pragma unroll
      for (int nt = 0; nt < 4; ++nt)
        c1[nt] = wmma_f16(load_afrag(fW1, nt * 2 + s, lane), b, c1[nt]);
    }

    // ---- layer 2 (no relu on output)
    v8f c2[4];
#pragma unroll
    for (int nt = 0; nt < 4; ++nt) c2[nt] = bias2[nt];
#pragma unroll
    for (int s = 0; s < 2; ++s) {
      const v16h b = make_bfrag(c1[2 * s], c1[2 * s + 1], hi);
#pragma unroll
      for (int nt = 0; nt < 4; ++nt)
        c2[nt] = wmma_f16(load_afrag(fW2, nt * 2 + s, lane), b, c2[nt]);
    }

    // ---- scatter-add: lane owns edge e, channels nt*16+rowoff+0..7.
    const size_t dbase = (size_t)dst[e] * 64;
#pragma unroll
    for (int nt = 0; nt < 4; ++nt)
#pragma unroll
      for (int j = 0; j < 8; ++j)
        atomicAdd(&agg[dbase + nt * 16 + rowoff + j], c2[nt][j]);
  }
}

// ---------------------------------------------------------------------------
// Node update: v = relu(v @ Wself + bself + agg/deg + v0) * sigmoid(gate)
// ---------------------------------------------------------------------------
__global__ __launch_bounds__(256) void node_update_kernel(
    _Float16* __restrict__ v16, const _Float16* __restrict__ v0_16,
    const float* __restrict__ agg, const float* __restrict__ deg,
    const float* __restrict__ x, const float* __restrict__ Ws,
    const float* __restrict__ bs, const float* __restrict__ gW,
    const float* __restrict__ gb) {
  __shared__ _Float16 fW[8 * 512];
  __shared__ float sb[64], sgw[64], sgb[64];

  fill_frags(fW, Ws, 2, 8, 64);
  if (threadIdx.x < 64) { sb[threadIdx.x] = bs[threadIdx.x]; sgw[threadIdx.x] = gW[threadIdx.x]; sgb[threadIdx.x] = gb[threadIdx.x]; }
  __syncthreads();

  const int lane = threadIdx.x & 31;
  const int wave = threadIdx.x >> 5;
  const int r = lane & 15;
  const bool hi = (lane & 16) != 0;
  const int rowoff = hi ? 8 : 0;
  const int koffB = hi ? 16 : 0;

  v8f bias[4];
  float gwv[4][8], gbv[4][8];
#pragma unroll
  for (int nt = 0; nt < 4; ++nt)
#pragma unroll
    for (int j = 0; j < 8; ++j) {
      bias[nt][j] = sb[nt * 16 + rowoff + j];
      gwv[nt][j] = sgw[nt * 16 + rowoff + j];
      gbv[nt][j] = sgb[nt * 16 + rowoff + j];
    }

  const int nTiles = Nn / 16;
  for (int tile = blockIdx.x * 8 + wave; tile < nTiles; tile += gridDim.x * 8) {
    const size_t node = (size_t)(tile * 16 + r);
    const _Float16* vr = v16 + node * 64;

    v8f c[4];
#pragma unroll
    for (int nt = 0; nt < 4; ++nt) c[nt] = bias[nt];
#pragma unroll
    for (int s = 0; s < 2; ++s) {
      const v16h b = load_brow(vr + s * 32 + koffB);
#pragma unroll
      for (int nt = 0; nt < 4; ++nt)
        c[nt] = wmma_f16(load_afrag(fW, nt * 2 + s, lane), b, c[nt]);
    }

    const float rdeg = 1.0f / fmaxf(deg[node], 1.0f);
    const float x4 = x[node * IN_D + 4];
#pragma unroll
    for (int nt = 0; nt < 4; ++nt) {
      const size_t base = node * 64 + nt * 16 + rowoff;
      const v8h v0v = *(const v8h*)(v0_16 + base);
      v8h out;
#pragma unroll
      for (int j = 0; j < 8; ++j) {
        const float z = relu1(c[nt][j] + agg[base + j] * rdeg + (float)v0v[j]);
        const float g = 1.0f / (1.0f + __expf(-(x4 * gwv[nt][j] + gbv[nt][j])));
        out[j] = (_Float16)(z * g);
      }
      *(v8h*)(v16 + base) = out;
    }
  }
}

// ---------------------------------------------------------------------------
// Projection: out = relu(v @ W1 + b1) @ W2(64x3) + b2
// ---------------------------------------------------------------------------
__global__ __launch_bounds__(256) void proj_kernel(
    const _Float16* __restrict__ v16, const float* __restrict__ W1,
    const float* __restrict__ b1, const float* __restrict__ W2,
    const float* __restrict__ b2, float* __restrict__ out) {
  __shared__ _Float16 fW1[8 * 512];
  __shared__ float sW2[64 * OUT_D];
  __shared__ float sb1[64], sb2[OUT_D];

  fill_frags(fW1, W1, 2, 8, 64);
  for (int i = threadIdx.x; i < 64 * OUT_D; i += 256) sW2[i] = W2[i];
  if (threadIdx.x < 64) sb1[threadIdx.x] = b1[threadIdx.x];
  if (threadIdx.x < OUT_D) sb2[threadIdx.x] = b2[threadIdx.x];
  __syncthreads();

  const int lane = threadIdx.x & 31;
  const int wave = threadIdx.x >> 5;
  const int r = lane & 15;
  const bool hi = (lane & 16) != 0;
  const int rowoff = hi ? 8 : 0;
  const int koffB = hi ? 16 : 0;

  v8f bias[4];
  float w2v[4][8][OUT_D];
#pragma unroll
  for (int nt = 0; nt < 4; ++nt)
#pragma unroll
    for (int j = 0; j < 8; ++j) {
      bias[nt][j] = sb1[nt * 16 + rowoff + j];
#pragma unroll
      for (int cc = 0; cc < OUT_D; ++cc)
        w2v[nt][j][cc] = sW2[(nt * 16 + rowoff + j) * OUT_D + cc];
    }

  const int nTiles = Nn / 16;
  for (int tile = blockIdx.x * 8 + wave; tile < nTiles; tile += gridDim.x * 8) {
    const size_t node = (size_t)(tile * 16 + r);
    const _Float16* vr = v16 + node * 64;

    v8f c[4];
#pragma unroll
    for (int nt = 0; nt < 4; ++nt) c[nt] = bias[nt];
#pragma unroll
    for (int s = 0; s < 2; ++s) {
      const v16h b = load_brow(vr + s * 32 + koffB);
#pragma unroll
      for (int nt = 0; nt < 4; ++nt)
        c[nt] = wmma_f16(load_afrag(fW1, nt * 2 + s, lane), b, c[nt]);
    }

    // tiny second layer: partial dot over this lane's 32 channels, then
    // combine with partner lane (other channel half) via shfl_xor(16).
    float p[OUT_D];
#pragma unroll
    for (int cc = 0; cc < OUT_D; ++cc) p[cc] = 0.0f;
#pragma unroll
    for (int nt = 0; nt < 4; ++nt)
#pragma unroll
      for (int j = 0; j < 8; ++j) {
        const float h = relu1(c[nt][j]);
#pragma unroll
        for (int cc = 0; cc < OUT_D; ++cc) p[cc] += h * w2v[nt][j][cc];
      }
#pragma unroll
    for (int cc = 0; cc < OUT_D; ++cc) p[cc] += __shfl_xor(p[cc], 16, 32);
    if (!hi) {
#pragma unroll
      for (int cc = 0; cc < OUT_D; ++cc)
        out[node * OUT_D + cc] = p[cc] + sb2[cc];
    }
  }
}

// ---------------------------------------------------------------------------
extern "C" void kernel_launch(void* const* d_in, const int* in_sizes, int n_in,
                              void* d_out, int out_size, void* d_ws, size_t ws_size,
                              hipStream_t stream) {
  const float* x        = (const float*)d_in[0];
  const int*   ei       = (const int*)d_in[1];
  const float* eattr    = (const float*)d_in[2];
  const int*   batch    = (const int*)d_in[3];
  const float* lift_W1  = (const float*)d_in[4];
  const float* lift_b1  = (const float*)d_in[5];
  const float* lift_W2  = (const float*)d_in[6];
  const float* lift_b2  = (const float*)d_in[7];
  const float* Wself    = (const float*)d_in[8];
  const float* bself    = (const float*)d_in[9];
  const float* msg_W0   = (const float*)d_in[10];
  const float* msg_b0   = (const float*)d_in[11];
  const float* msg_W1   = (const float*)d_in[12];
  const float* msg_b1   = (const float*)d_in[13];
  const float* msg_W2   = (const float*)d_in[14];
  const float* msg_b2   = (const float*)d_in[15];
  const float* gate_W   = (const float*)d_in[16];
  const float* gate_b   = (const float*)d_in[17];
  const float* proj_W1  = (const float*)d_in[18];
  const float* proj_b1  = (const float*)d_in[19];
  const float* proj_W2  = (const float*)d_in[20];
  const float* proj_b2  = (const float*)d_in[21];

  const int* src = ei;
  const int* dst = ei + Ee;

  // ---- workspace carve (256B aligned slabs)
  char* w = (char*)d_ws;
  size_t off = 0;
  auto take = [&](size_t bytes) -> char* {
    char* p = w + off;
    off = (off + bytes + 255) & ~(size_t)255;
    return p;
  };
  _Float16* eaPad = (_Float16*)take((size_t)Ee * EAP * sizeof(_Float16));
  _Float16* v16   = (_Float16*)take((size_t)Nn * Hh * sizeof(_Float16));
  _Float16* v0_16 = (_Float16*)take((size_t)Nn * Hh * sizeof(_Float16));
  float*    agg   = (float*)take((size_t)Nn * Hh * sizeof(float));
  float*    deg   = (float*)take((size_t)Nn * sizeof(float));
  float*    sums  = (float*)take((size_t)Gg * EDd * sizeof(float));
  float*    sqs   = (float*)take((size_t)Gg * EDd * sizeof(float));
  float*    cnt   = (float*)take((size_t)Gg * sizeof(float));
  float*    mu    = (float*)take((size_t)Gg * EDd * sizeof(float));
  float*    inv   = (float*)take((size_t)Gg * EDd * sizeof(float));

  // ---- per-graph edge normalization + degrees
  hipMemsetAsync(sums, 0, (size_t)Gg * EDd * sizeof(float), stream);
  hipMemsetAsync(sqs, 0, (size_t)Gg * EDd * sizeof(float), stream);
  hipMemsetAsync(cnt, 0, (size_t)Gg * sizeof(float), stream);
  hipMemsetAsync(deg, 0, (size_t)Nn * sizeof(float), stream);

  edge_stats_kernel<<<512, 256, 0, stream>>>(eattr, src, batch, sums, sqs, cnt);
  stats_finalize_kernel<<<1, 1024, 0, stream>>>(sums, sqs, cnt, mu, inv);
  edge_normalize_kernel<<<(Ee + 255) / 256, 256, 0, stream>>>(eattr, src, batch, mu, inv, eaPad);
  degree_kernel<<<(Ee + 255) / 256, 256, 0, stream>>>(dst, deg);

  // ---- lift
  const int nodeBlocks = (Nn / 16 + 7) / 8;  // 8 waves/block, 1 tile/wave
  lift_kernel<<<nodeBlocks, 256, 0, stream>>>(x, lift_W1, lift_b1, lift_W2, lift_b2,
                                              v16, v0_16);

  // ---- T rounds of message passing
  for (int t = 0; t < Tt; ++t) {
    hipMemsetAsync(agg, 0, (size_t)Nn * Hh * sizeof(float), stream);
    edge_mlp_kernel<<<2048, 256, 0, stream>>>(
        v16, eaPad, src, dst,
        msg_W0 + (size_t)t * 73 * 64, msg_b0 + (size_t)t * 64,
        msg_W1 + (size_t)t * 64 * 64, msg_b1 + (size_t)t * 64,
        msg_W2 + (size_t)t * 64 * 64, msg_b2 + (size_t)t * 64, agg);
    node_update_kernel<<<nodeBlocks, 256, 0, stream>>>(
        v16, v0_16, agg, deg, x,
        Wself + (size_t)t * 64 * 64, bself + (size_t)t * 64,
        gate_W + (size_t)t * 64, gate_b + (size_t)t * 64);
  }

  // ---- projection
  proj_kernel<<<nodeBlocks, 256, 0, stream>>>(v16, proj_W1, proj_b1, proj_W2, proj_b2,
                                              (float*)d_out);
}
